// SparselyGatedMoE_51281909514341
// MI455X (gfx1250) — compile-verified, matrix operand order checked by reading, not run on hardware
//
#include <hip/hip_runtime.h>
#include <cstdint>
#include <cstddef>

// Problem dims (fixed by the reference)
#define BB   512
#define SS   64
#define DD   512
#define EE   8
#define LL   4
#define HH   256
#define NHH  4
#define OO   128
#define HDD  64
#define FFF  1024

typedef __attribute__((ext_vector_type(16))) _Float16 v16h;
typedef __attribute__((ext_vector_type(8)))  float    v8f;

union U16h { v16h v; uint4 u[2]; _Float16 h[16]; };
union U8h  { uint4 u; _Float16 h[8]; };

// ---------------------------------------------------------------------------
// CDNA5 async global->LDS copy (ASYNCcnt) with sync fallback.
// Builtin parameter type (from hipcc diagnostic): v4i32* (vector of 4 ints).
// ---------------------------------------------------------------------------
#if defined(__has_builtin)
#if __has_builtin(__builtin_amdgcn_global_load_async_to_lds_b128)
#define HAVE_ASYNC_LDS 1
#endif
#endif
#ifndef HAVE_ASYNC_LDS
#define HAVE_ASYNC_LDS 0
#endif

#if HAVE_ASYNC_LDS
typedef __attribute__((vector_size(16))) int v4i_t;
__device__ __forceinline__ void async_cp16(const void* g, void* l) {
  __builtin_amdgcn_global_load_async_to_lds_b128(
      (__attribute__((address_space(1))) v4i_t*)(g),
      (__attribute__((address_space(3))) v4i_t*)(l), 0, 0);
}
__device__ __forceinline__ void wait_async0() {
#if __has_builtin(__builtin_amdgcn_s_wait_asynccnt)
  __builtin_amdgcn_s_wait_asynccnt(0);
#else
  asm volatile("s_wait_asynccnt 0" ::: "memory");
#endif
}
#endif

// ---------------------------------------------------------------------------
// Weight pre-swizzle: f32 [K,N] row-major  ->  f16 WMMA B-fragment order:
//   frag(kt=k>>5, ncg=n>>4) is 512 halves: [lane][e], lane = (n&15)|(((k>>3)&1)<<4),
//   e = (((k>>4)&1)<<3)|(k&7).  One 16B chunk = 8 consecutive k for fixed lane.
// Batched over gridDim.y (matrices are contiguous).
// ---------------------------------------------------------------------------
__global__ __launch_bounds__(256) void swz_kernel(
    const float* __restrict__ Win, _Float16* __restrict__ out,
    int N, int chunks)   // chunks = K*N/8 per matrix
{
  const size_t matOff = (size_t)blockIdx.y * chunks * 8;
  const float* W = Win + matOff;
  _Float16* o = out + matOff;
  int c = blockIdx.x * 256 + threadIdx.x;
  if (c >= chunks) return;
  int fragIdx = c >> 6;
  int rem = c & 63;
  int l = rem >> 1;
  int G = rem & 1;
  int ntiles = N >> 4;
  int ktile = fragIdx / ntiles;
  int ncg = fragIdx - ktile * ntiles;
  int n = (ncg << 4) + (l & 15);
  int k0 = (ktile << 5) + (G << 4) + ((l >> 4) << 3);
  U8h v;
#pragma unroll
  for (int j = 0; j < 8; ++j) v.h[j] = (_Float16)W[(size_t)(k0 + j) * N + n];
  *(uint4*)(o + (size_t)c * 8) = v.u;
}

// ---------------------------------------------------------------------------
// Tiled WMMA GEMM:  C[M,N] (op)= act(A[M,K] @ W[K,N] + bias)
//   A: f16 (AF16=1) or f32 (AF16=0), row-major.
//   Wsw: f16 weights pre-swizzled in B-fragment order (see swz_kernel).
//   OMODE: 0 = store f32, 1 = C += (f32), 2 = store f16,
//          3 = C += rowscale[row*rs_stride] * val  (f32)
// Block tile 128x128, 8 waves, each wave 2x4 wmma tiles (8 wmma / K-step).
// Double-buffered LDS; staging uses async global->LDS b128 when available.
// ---------------------------------------------------------------------------
template<bool AF16, int OMODE, bool RELU>
__global__ __launch_bounds__(256) void gemm_wmma(
    const void* __restrict__ Av, const _Float16* __restrict__ Wsw,
    const float* __restrict__ bias, void* __restrict__ Cv,
    const float* __restrict__ rowscale, int rs_stride,
    int M, int N, int Kd)
{
  __shared__ __align__(32) _Float16 sA[2][8][32][16];  // 8 row-groups of 16 rows
  __shared__ __align__(32) _Float16 sB[2][8][32][16];  // 8 col-groups of 16 cols
  const int tid  = threadIdx.x;
  const int lane = tid & 31;
  const int wave = tid >> 5;
  const int wr = wave & 3, wc = wave >> 2;
  const int m0 = blockIdx.y * 128;
  const int n0 = blockIdx.x * 128;
  const _Float16* A16 = (const _Float16*)Av;
  const float*    A32 = (const float*)Av;
  const int ntiles = N >> 4;
  const int T = Kd >> 5;

  v8f acc[2][4] = {};
  uint4 pa[2];

  // chunk decode: c = tid + cc*256 -> (grp = c>>6, l = (c>>1)&31, G = c&1)
  auto aSrcOff = [&](int cc, int kt) -> size_t {
    int c = tid + cc * 256;
    int g = c >> 6, l = (c >> 1) & 31, G = c & 1;
    int r  = m0 + g * 16 + (l & 15);
    int kk = kt + G * 16 + ((l >> 4) << 3);
    return (size_t)r * Kd + kk;
  };
  auto aDst = [&](int cc, int buf) -> _Float16* {
    int c = tid + cc * 256;
    int g = c >> 6, l = (c >> 1) & 31, G = c & 1;
    return &sA[buf][g][l][G * 8];
  };
  auto bSrc = [&](int cc, int kt) -> const _Float16* {
    int c = tid + cc * 256;
    int cg = c >> 6, l = (c >> 1) & 31, G = c & 1;
    size_t frag = (size_t)(kt >> 5) * ntiles + (n0 >> 4) + cg;
    return Wsw + frag * 512 + l * 16 + G * 8;
  };
  auto bDst = [&](int cc, int buf) -> _Float16* {
    int c = tid + cc * 256;
    int cg = c >> 6, l = (c >> 1) & 31, G = c & 1;
    return &sB[buf][cg][l][G * 8];
  };
  auto ldA = [&](int kt) {   // into registers (sync path / f32 conversion path)
#pragma unroll
    for (int cc = 0; cc < 2; ++cc) {
      if (AF16) {
        pa[cc] = *(const uint4*)(A16 + aSrcOff(cc, kt));
      } else {
        const float* ap = A32 + aSrcOff(cc, kt);
        float4 f0 = *(const float4*)ap;
        float4 f1 = *(const float4*)(ap + 4);
        U8h v;
        v.h[0]=(_Float16)f0.x; v.h[1]=(_Float16)f0.y;
        v.h[2]=(_Float16)f0.z; v.h[3]=(_Float16)f0.w;
        v.h[4]=(_Float16)f1.x; v.h[5]=(_Float16)f1.y;
        v.h[6]=(_Float16)f1.z; v.h[7]=(_Float16)f1.w;
        pa[cc] = v.u;
      }
    }
  };
  auto stA = [&](int buf) {
#pragma unroll
    for (int cc = 0; cc < 2; ++cc) *(uint4*)aDst(cc, buf) = pa[cc];
  };

#if HAVE_ASYNC_LDS
  // prologue: issue tile 0
#pragma unroll
  for (int cc = 0; cc < 2; ++cc) async_cp16(bSrc(cc, 0), bDst(cc, 0));
  if (AF16) {
#pragma unroll
    for (int cc = 0; cc < 2; ++cc) async_cp16(A16 + aSrcOff(cc, 0), aDst(cc, 0));
  } else {
    ldA(0);
  }
#else
  uint4 pb[2];
  auto ldB = [&](int kt) {
#pragma unroll
    for (int cc = 0; cc < 2; ++cc) pb[cc] = *(const uint4*)bSrc(cc, kt);
  };
  auto stB = [&](int buf) {
#pragma unroll
    for (int cc = 0; cc < 2; ++cc) *(uint4*)bDst(cc, buf) = pb[cc];
  };
  ldA(0); ldB(0);
#endif

  for (int t = 0; t < T; ++t) {
    const int buf = t & 1;
    const int kt1 = (t + 1) << 5;
#if HAVE_ASYNC_LDS
    if (!AF16) stA(buf);
    wait_async0();
    __syncthreads();
    if (t + 1 < T) {
#pragma unroll
      for (int cc = 0; cc < 2; ++cc) async_cp16(bSrc(cc, kt1), bDst(cc, buf ^ 1));
      if (AF16) {
#pragma unroll
        for (int cc = 0; cc < 2; ++cc) async_cp16(A16 + aSrcOff(cc, kt1), aDst(cc, buf ^ 1));
      } else {
        ldA(kt1);
      }
    }
#else
    stA(buf); stB(buf);
    __syncthreads();
    if (t + 1 < T) { ldA(kt1); ldB(kt1); }
#endif
    // ---- compute: 2 A-frags x 4 B-frags = 8 wmma ----
    v16h a0 = *(const v16h*)sA[buf][wr * 2 + 0][lane];
    v16h a1 = *(const v16h*)sA[buf][wr * 2 + 1][lane];
    v16h b0 = *(const v16h*)sB[buf][wc * 4 + 0][lane];
    v16h b1 = *(const v16h*)sB[buf][wc * 4 + 1][lane];
    v16h b2 = *(const v16h*)sB[buf][wc * 4 + 2][lane];
    v16h b3 = *(const v16h*)sB[buf][wc * 4 + 3][lane];
    acc[0][0] = __builtin_amdgcn_wmma_f32_16x16x32_f16(false, a0, false, b0, (short)0, acc[0][0], false, false);
    acc[0][1] = __builtin_amdgcn_wmma_f32_16x16x32_f16(false, a0, false, b1, (short)0, acc[0][1], false, false);
    acc[0][2] = __builtin_amdgcn_wmma_f32_16x16x32_f16(false, a0, false, b2, (short)0, acc[0][2], false, false);
    acc[0][3] = __builtin_amdgcn_wmma_f32_16x16x32_f16(false, a0, false, b3, (short)0, acc[0][3], false, false);
    acc[1][0] = __builtin_amdgcn_wmma_f32_16x16x32_f16(false, a1, false, b0, (short)0, acc[1][0], false, false);
    acc[1][1] = __builtin_amdgcn_wmma_f32_16x16x32_f16(false, a1, false, b1, (short)0, acc[1][1], false, false);
    acc[1][2] = __builtin_amdgcn_wmma_f32_16x16x32_f16(false, a1, false, b2, (short)0, acc[1][2], false, false);
    acc[1][3] = __builtin_amdgcn_wmma_f32_16x16x32_f16(false, a1, false, b3, (short)0, acc[1][3], false, false);
  }

  // ---- epilogue: C layout = lane holds col (lane&15), rows i + 8*(lane>>4) ----
  float*    C32 = (float*)Cv;
  _Float16* C16 = (_Float16*)Cv;
#pragma unroll
  for (int tm = 0; tm < 2; ++tm) {
#pragma unroll
    for (int tn = 0; tn < 4; ++tn) {
      int g  = wr * 2 + tm;
      int cg = wc * 4 + tn;
      int n  = n0 + cg * 16 + (lane & 15);
      float bb = bias ? bias[n] : 0.0f;
      int rbase = m0 + g * 16 + ((lane >> 4) << 3);
#pragma unroll
      for (int i = 0; i < 8; ++i) {
        float val = acc[tm][tn][i] + bb;
        if (RELU) val = fmaxf(val, 0.0f);
        size_t offc = (size_t)(rbase + i) * N + n;
        if (OMODE == 0)      C32[offc] = val;
        else if (OMODE == 1) C32[offc] += val;
        else if (OMODE == 2) C16[offc] = (_Float16)val;
        else                 C32[offc] += rowscale[(size_t)(rbase + i) * rs_stride] * val;
      }
    }
  }
}

// ---------------------------------------------------------------------------
// WMMA attention: one block = one (batch b, head hh). S=64, HD=64.
// ---------------------------------------------------------------------------
__global__ __launch_bounds__(128) void attn_kernel(
    const _Float16* __restrict__ q, const _Float16* __restrict__ k,
    const _Float16* __restrict__ v, _Float16* __restrict__ o)
{
  __shared__ float sS[64][65];
  __shared__ __align__(32) _Float16 pA[4][2][32][16];
  __shared__ __align__(32) _Float16 vB[4][2][32][16];

  const int tid  = threadIdx.x;
  const int lane = tid & 31;
  const int w    = tid >> 5;
  const int b    = blockIdx.x / NHH;
  const int hh   = blockIdx.x % NHH;
  const size_t base = (size_t)b * SS * HH + (size_t)hh * HDD;
  const _Float16* qp = q + base;
  const _Float16* kp = k + base;
  const _Float16* vp = v + base;
  _Float16*       op = o + base;
  const int hi8  = (lane >> 4) << 3;
  const int lmod = lane & 15;

  // ---- phase 1: scores ----
  v8f accS[4] = {};
  const int srow = 16 * w + lmod;
#pragma unroll
  for (int kk = 0; kk < 2; ++kk) {
    U16h af;
    const _Float16* ab = qp + (size_t)srow * HH + kk * 32;
    af.u[0] = *(const uint4*)(ab + hi8);
    af.u[1] = *(const uint4*)(ab + 16 + hi8);
#pragma unroll
    for (int j = 0; j < 4; ++j) {
      U16h bf;
      const _Float16* bbp = kp + (size_t)(16 * j + lmod) * HH + kk * 32;
      bf.u[0] = *(const uint4*)(bbp + hi8);
      bf.u[1] = *(const uint4*)(bbp + 16 + hi8);
      accS[j] = __builtin_amdgcn_wmma_f32_16x16x32_f16(false, af.v, false, bf.v, (short)0, accS[j], false, false);
    }
  }
#pragma unroll
  for (int j = 0; j < 4; ++j)
#pragma unroll
    for (int i = 0; i < 8; ++i)
      sS[16 * w + i + hi8][16 * j + lmod] = accS[j][i] * 0.125f;
  __syncthreads();

  // ---- phase 2a: stage V^T in B-fragment order ----
  for (int task = tid; task < 512; task += 128) {
    int t  = task >> 3;
    int d0 = (task & 7) * 8;
    int j   = t >> 5;
    int kkp = t & 31;
    int lhi = ((kkp >> 3) & 1) << 4;
    int e   = (((kkp >> 4) & 1) << 3) | (kkp & 7);
    const _Float16* vrow = vp + (size_t)t * HH + d0;
#pragma unroll
    for (int ii = 0; ii < 8; ++ii) {
      int d = d0 + ii;
      vB[d >> 4][j][(d & 15) | lhi][e] = vrow[ii];
    }
  }
  // ---- phase 2b: softmax ----
  if (tid < 64) {
    int r = tid;
    float mx = -1e30f;
#pragma unroll
    for (int t = 0; t < 64; ++t) mx = fmaxf(mx, sS[r][t]);
    float p[64];
    float sum = 0.0f;
#pragma unroll
    for (int t = 0; t < 64; ++t) { p[t] = __expf(sS[r][t] - mx); sum += p[t]; }
    float inv = 1.0f / sum;
    int rg = r >> 4;
#pragma unroll
    for (int t = 0; t < 64; ++t) {
      int j   = t >> 5;
      int kkp = t & 31;
      int lhi = ((kkp >> 3) & 1) << 4;
      int e   = (((kkp >> 4) & 1) << 3) | (kkp & 7);
      pA[rg][j][(r & 15) | lhi][e] = (_Float16)(p[t] * inv);
    }
  }
  __syncthreads();

  // ---- phase 3: O = P @ V ----
#pragma unroll
  for (int cg = 0; cg < 4; ++cg) {
    v8f accO = {};
#pragma unroll
    for (int j = 0; j < 2; ++j) {
      v16h ap  = *(const v16h*)pA[w][j][lane];
      v16h bvv = *(const v16h*)vB[cg][j][lane];
      accO = __builtin_amdgcn_wmma_f32_16x16x32_f16(false, ap, false, bvv, (short)0, accO, false, false);
    }
    int d = 16 * cg + lmod;
#pragma unroll
    for (int i = 0; i < 8; ++i)
      op[(size_t)(16 * w + i + hi8) * HH + d] = (_Float16)accO[i];
  }
}

// ---------------------------------------------------------------------------
// LayerNorm over H=256: one wave per row, wave32 shuffle reduction, no LDS.
// grid = rows/8, block = 256 (8 waves).
// ---------------------------------------------------------------------------
__global__ __launch_bounds__(256) void ln256_kernel(
    const float* __restrict__ x, const float* __restrict__ g,
    const float* __restrict__ bvec, _Float16* __restrict__ out)
{
  const int lane = threadIdx.x & 31;
  const int r = blockIdx.x * 8 + (threadIdx.x >> 5);
  const float* xp = x + (size_t)r * HH + lane * 8;
  float4 f0 = *(const float4*)xp;
  float4 f1 = *(const float4*)(xp + 4);
  float xv[8] = {f0.x, f0.y, f0.z, f0.w, f1.x, f1.y, f1.z, f1.w};
  float s = 0.0f, ss = 0.0f;
#pragma unroll
  for (int j = 0; j < 8; ++j) { s += xv[j]; ss += xv[j] * xv[j]; }
#pragma unroll
  for (int m = 16; m > 0; m >>= 1) {
    s  += __shfl_xor(s, m, 32);
    ss += __shfl_xor(ss, m, 32);
  }
  float mean = s * (1.0f / HH);
  float var  = ss * (1.0f / HH) - mean * mean;
  float rstd = rsqrtf(var + 1e-5f);
  const float* gp = g + lane * 8;
  const float* bp = bvec + lane * 8;
  float4 g0 = *(const float4*)gp, g1 = *(const float4*)(gp + 4);
  float4 c0 = *(const float4*)bp, c1 = *(const float4*)(bp + 4);
  float gv[8] = {g0.x, g0.y, g0.z, g0.w, g1.x, g1.y, g1.z, g1.w};
  float bw[8] = {c0.x, c0.y, c0.z, c0.w, c1.x, c1.y, c1.z, c1.w};
  U8h o;
#pragma unroll
  for (int j = 0; j < 8; ++j) o.h[j] = (_Float16)((xv[j] - mean) * rstd * gv[j] + bw[j]);
  *(uint4*)(out + (size_t)r * HH + lane * 8) = o.u;
}

// mean over S + final LN, writes f16 pooled [B,H]
__global__ __launch_bounds__(256) void pool_lnf_kernel(
    const float* __restrict__ h, const float* __restrict__ g,
    const float* __restrict__ bvec, _Float16* __restrict__ out)
{
  __shared__ float s1[256], s2[256];
  int b = blockIdx.x, t = threadIdx.x;
  const float* hp = h + (size_t)b * SS * HH + t;
  float acc = 0.0f;
  for (int s = 0; s < SS; ++s) acc += hp[(size_t)s * HH];
  float val = acc * (1.0f / SS);
  s1[t] = val; s2[t] = val * val;
  __syncthreads();
  for (int off = 128; off > 0; off >>= 1) {
    if (t < off) { s1[t] += s1[t + off]; s2[t] += s2[t + off]; }
    __syncthreads();
  }
  float m  = s1[0] * (1.0f / HH);
  float va = s2[0] * (1.0f / HH) - m * m;
  float rs = rsqrtf(va + 1e-5f);
  out[(size_t)b * HH + t] = (_Float16)((val - m) * rs * g[t] + bvec[t]);
}

// gate_in[b,d] = mean over s of x[b,s,d]
__global__ __launch_bounds__(256) void gate_mean_kernel(
    const float* __restrict__ x, float* __restrict__ gi)
{
  int idx = blockIdx.x * 256 + threadIdx.x;
  int b = idx >> 9;
  int d = idx & (DD - 1);
  const float* xp = x + (size_t)b * SS * DD + d;
  float s = 0.0f;
  for (int t = 0; t < SS; ++t) s += xp[(size_t)t * DD];
  gi[idx] = s * (1.0f / SS);
}

// logits + top-2 softmax -> dense combine weights [B,E]
__global__ __launch_bounds__(256) void gate_topk_kernel(
    const float* __restrict__ gi, const float* __restrict__ gW,
    const float* __restrict__ gb, float* __restrict__ wdense)
{
  __shared__ float sx[DD];
  __shared__ float lg[EE];
  int b = blockIdx.x, t = threadIdx.x;
  int lane = t & 31, w = t >> 5;
  sx[t]       = gi[(size_t)b * DD + t];
  sx[t + 256] = gi[(size_t)b * DD + t + 256];
  __syncthreads();
  float a = 0.0f;
  for (int d = lane; d < DD; d += 32) a += sx[d] * gW[(size_t)d * EE + w];
  for (int off = 16; off > 0; off >>= 1) a += __shfl_down(a, off, 32);
  if (lane == 0) lg[w] = a + gb[w];
  __syncthreads();
  if (t == 0) {
    int i0 = 0;
    for (int e = 1; e < EE; ++e) if (lg[e] > lg[i0]) i0 = e;
    int i1 = -1;
    for (int e = 0; e < EE; ++e) if (e != i0 && (i1 < 0 || lg[e] > lg[i1])) i1 = e;
    float e1  = __expf(lg[i1] - lg[i0]);
    float inv = 1.0f / (1.0f + e1);
    for (int e = 0; e < EE; ++e) wdense[b * EE + e] = 0.0f;
    wdense[b * EE + i0] = inv;
    wdense[b * EE + i1] = e1 * inv;
  }
}

__global__ void zero_kernel(float* p, int n)
{
  int i = blockIdx.x * 256 + threadIdx.x;
  if (i < n) p[i] = 0.0f;
}

// ---------------------------------------------------------------------------
// Orchestration.
// ---------------------------------------------------------------------------
extern "C" void kernel_launch(void* const* d_in, const int* in_sizes, int n_in,
                              void* d_out, int out_size, void* d_ws, size_t ws_size,
                              hipStream_t stream)
{
  (void)in_sizes; (void)n_in; (void)out_size; (void)ws_size;
  const float* x      = (const float*)d_in[0];
  const float* gate_W = (const float*)d_in[1];
  const float* gate_b = (const float*)d_in[2];
  const float* emb_W  = (const float*)d_in[3];
  const float* emb_b  = (const float*)d_in[4];
  const float* ln1_g  = (const float*)d_in[5];
  const float* ln1_b  = (const float*)d_in[6];
  const float* Wq     = (const float*)d_in[7];
  const float* bq     = (const float*)d_in[8];
  const float* Wk     = (const float*)d_in[9];
  const float* bk     = (const float*)d_in[10];
  const float* Wv     = (const float*)d_in[11];
  const float* bvp    = (const float*)d_in[12];
  const float* Wo     = (const float*)d_in[13];
  const float* bo     = (const float*)d_in[14];
  const float* ln2_g  = (const float*)d_in[15];
  const float* ln2_b  = (const float*)d_in[16];
  const float* W1     = (const float*)d_in[17];
  const float* b1     = (const float*)d_in[18];
  const float* W2     = (const float*)d_in[19];
  const float* b2     = (const float*)d_in[20];
  const float* lnf_g  = (const float*)d_in[21];
  const float* lnf_b  = (const float*)d_in[22];
  const float* head_W = (const float*)d_in[23];
  const float* head_b = (const float*)d_in[24];

  char* ws = (char*)d_ws;
  size_t off = 0;
  auto take = [&](size_t bytes) {
    char* p = ws + off;
    off = (off + bytes + 255) & ~(size_t)255;
    return p;
  };
  float*    h    = (float*)   take((size_t)BB * SS * HH  * 4);
  _Float16* a16  = (_Float16*)take((size_t)BB * SS * HH  * 2);
  _Float16* q16  = (_Float16*)take((size_t)BB * SS * HH  * 2);
  _Float16* k16  = (_Float16*)take((size_t)BB * SS * HH  * 2);
  _Float16* v16  = (_Float16*)take((size_t)BB * SS * HH  * 2);
  _Float16* t16  = (_Float16*)take((size_t)BB * SS * FFF * 2);
  float*    gi   = (float*)   take((size_t)BB * DD * 4);
  float*    wden = (float*)   take((size_t)BB * EE * 4);
  _Float16* p16  = (_Float16*)take((size_t)BB * HH * 2);
  // pre-swizzled f16 weights
  _Float16* swzEmb  = (_Float16*)take((size_t)EE * DD * HH * 2);
  _Float16* swzQ    = (_Float16*)take((size_t)EE * LL * HH * HH * 2);
  _Float16* swzK    = (_Float16*)take((size_t)EE * LL * HH * HH * 2);
  _Float16* swzV    = (_Float16*)take((size_t)EE * LL * HH * HH * 2);
  _Float16* swzO    = (_Float16*)take((size_t)EE * LL * HH * HH * 2);
  _Float16* swzW1   = (_Float16*)take((size_t)EE * LL * HH * FFF * 2);
  _Float16* swzW2   = (_Float16*)take((size_t)EE * LL * FFF * HH * 2);
  _Float16* swzHead = (_Float16*)take((size_t)EE * HH * OO * 2);

  float* out = (float*)d_out;
  const int Mtok = BB * SS;   // 32768 token rows

  // ---- one-time (per call) weight conversion + fragment swizzle ----
  {
    int ch;
    ch = DD * HH / 8;
    swz_kernel<<<dim3(ch / 256, EE), 256, 0, stream>>>(emb_W, swzEmb, HH, ch);
    ch = HH * HH / 8;
    swz_kernel<<<dim3(ch / 256, EE * LL), 256, 0, stream>>>(Wq, swzQ, HH, ch);
    swz_kernel<<<dim3(ch / 256, EE * LL), 256, 0, stream>>>(Wk, swzK, HH, ch);
    swz_kernel<<<dim3(ch / 256, EE * LL), 256, 0, stream>>>(Wv, swzV, HH, ch);
    swz_kernel<<<dim3(ch / 256, EE * LL), 256, 0, stream>>>(Wo, swzO, HH, ch);
    ch = HH * FFF / 8;
    swz_kernel<<<dim3(ch / 256, EE * LL), 256, 0, stream>>>(W1, swzW1, FFF, ch);
    ch = FFF * HH / 8;
    swz_kernel<<<dim3(ch / 256, EE * LL), 256, 0, stream>>>(W2, swzW2, HH, ch);
    ch = HH * OO / 8;
    swz_kernel<<<dim3(ch / 256, EE), 256, 0, stream>>>(head_W, swzHead, OO, ch);
  }

  zero_kernel<<<(BB * OO + 255) / 256, 256, 0, stream>>>(out, BB * OO);
  gate_mean_kernel<<<(BB * DD) / 256, 256, 0, stream>>>(x, gi);
  gate_topk_kernel<<<BB, 256, 0, stream>>>(gi, gate_W, gate_b, wden);

  for (int e = 0; e < EE; ++e) {
    // embedding: h = x @ emb_W[e] + emb_b[e]   (A is f32)
    gemm_wmma<false, 0, false><<<dim3(HH / 128, Mtok / 128), 256, 0, stream>>>(
        x, swzEmb + (size_t)e * DD * HH, emb_b + (size_t)e * HH, h,
        nullptr, 0, Mtok, HH, DD);

    for (int l = 0; l < LL; ++l) {
      size_t wmo = (size_t)e * LL + l;
      ln256_kernel<<<Mtok / 8, 256, 0, stream>>>(h, ln1_g + wmo * HH, ln1_b + wmo * HH, a16);
      gemm_wmma<true, 2, false><<<dim3(HH / 128, Mtok / 128), 256, 0, stream>>>(
          a16, swzQ + wmo * HH * HH, bq + wmo * HH, q16, nullptr, 0, Mtok, HH, HH);
      gemm_wmma<true, 2, false><<<dim3(HH / 128, Mtok / 128), 256, 0, stream>>>(
          a16, swzK + wmo * HH * HH, bk + wmo * HH, k16, nullptr, 0, Mtok, HH, HH);
      gemm_wmma<true, 2, false><<<dim3(HH / 128, Mtok / 128), 256, 0, stream>>>(
          a16, swzV + wmo * HH * HH, bvp + wmo * HH, v16, nullptr, 0, Mtok, HH, HH);
      attn_kernel<<<BB * NHH, 128, 0, stream>>>(q16, k16, v16, a16);  // a16 <- attn out
      gemm_wmma<true, 1, false><<<dim3(HH / 128, Mtok / 128), 256, 0, stream>>>(
          a16, swzO + wmo * HH * HH, bo + wmo * HH, h, nullptr, 0, Mtok, HH, HH);
      ln256_kernel<<<Mtok / 8, 256, 0, stream>>>(h, ln2_g + wmo * HH, ln2_b + wmo * HH, q16);
      gemm_wmma<true, 2, true><<<dim3(FFF / 128, Mtok / 128), 256, 0, stream>>>(
          q16, swzW1 + wmo * HH * FFF, b1 + wmo * FFF, t16, nullptr, 0, Mtok, FFF, HH);
      gemm_wmma<true, 1, false><<<dim3(HH / 128, Mtok / 128), 256, 0, stream>>>(
          t16, swzW2 + wmo * FFF * HH, b2 + wmo * HH, h, nullptr, 0, Mtok, HH, FFF);
    }
    pool_lnf_kernel<<<BB, 256, 0, stream>>>(h, lnf_g + (size_t)e * HH, lnf_b + (size_t)e * HH, p16);
    gemm_wmma<true, 3, false><<<dim3(OO / 128, BB / 128), 256, 0, stream>>>(
        p16, swzHead + (size_t)e * HH * OO, head_b + (size_t)e * OO, out,
        wden + e, EE, BB, OO, HH);
  }
}
